// CosineTripletLoss_46531675684978
// MI455X (gfx1250) — matrix operand order repaired and checked by reading, not compiled, not run on hardware
//
#include <hip/hip_runtime.h>

#define BDIM 8192
#define DDIM 1024
#define MARGIN 0.05f
#define SEMI_HARD 0.05f

typedef __attribute__((ext_vector_type(16))) __bf16 v16bf;
typedef __attribute__((ext_vector_type(8)))  float  v8f;
typedef __attribute__((ext_vector_type(4)))  unsigned int v4u;
typedef __attribute__((ext_vector_type(8)))  unsigned int v8u;

// LDS tile: 128 rows x 32 bf16 (64B data) padded by TDM to 80B stride (20-bank
// stride -> conflict-free ds_load_b128). 128*80 = 10240B per buffer.
#define TILE_U4 640   // 10240 / 16 bytes

// ---------------- helpers ----------------

__device__ __forceinline__ unsigned short f2bf(float f) {
    unsigned u = __float_as_uint(f);
    unsigned r = u + 0x7FFFu + ((u >> 16) & 1u);
    return (unsigned short)(r >> 16);
}

// Monotone packing of (value, col) for atomicMax; equal value -> smaller col wins.
__device__ __forceinline__ unsigned long long pack_max(float v, unsigned idx) {
    unsigned ub = __float_as_uint(v);
    ub = (ub & 0x80000000u) ? ~ub : (ub | 0x80000000u);
    return ((unsigned long long)ub << 32) | (unsigned long long)(0xFFFFFFFFu - idx);
}

__device__ __forceinline__ unsigned lds_off(const void* p) {
    // flat LDS aperture: addr[31:0] is the LDS offset
    return (unsigned)(unsigned long long)(uintptr_t)p;
}

// Issue one TDM 2D tile load: 128 rows x 32 bf16 from row-major [8192][1024]
// bf16 matrix into LDS, padding +16B after every 64B row (80B LDS stride).
__device__ __forceinline__ void tdm_tile_load(unsigned lds_addr, const unsigned short* gptr) {
    unsigned long long ga = (unsigned long long)(uintptr_t)gptr;
    v4u g0;
    g0.x = 1u;                                   // count=1 (valid user descriptor)
    g0.y = lds_addr;                             // LDS byte address
    g0.z = (unsigned)(ga & 0xFFFFFFFFull);       // global_addr[31:0]
    g0.w = (unsigned)((ga >> 32) & 0x01FFFFFFull) | (2u << 30);  // addr[56:32] | type=2

    v8u g1;
    // data_size=1(2B) | pad_enable | pad_interval=3(16 dwords) | pad_amount=3(4 dwords)
    g1.s0 = (1u << 16) | (1u << 20) | (3u << 22) | (3u << 25);
    g1.s1 = (unsigned)(DDIM & 0xFFFF) << 16;                       // tensor_dim0 lo16
    g1.s2 = (DDIM >> 16) | ((unsigned)(BDIM & 0xFFFF) << 16);      // dim0 hi | dim1 lo
    g1.s3 = (BDIM >> 16) | (32u << 16);                            // dim1 hi | tile_dim0=32
    g1.s4 = 128u;                                                  // tile_dim1=128
    g1.s5 = (unsigned)DDIM;                                        // tensor_dim0_stride
    g1.s6 = 0u;
    g1.s7 = 0u;

    asm volatile("tensor_load_to_lds %0, %1" :: "s"(g0), "s"(g1) : "memory");
}

// ---------------- kernel 1: init packed argmax cells ----------------

__global__ void ctl_init_kernel(unsigned long long* __restrict__ packed) {
    int i = blockIdx.x * blockDim.x + threadIdx.x;
    if (i < BDIM) packed[i] = pack_max(-1.0f, 0u);
}

// ---------------- kernel 2: positive_sim (fp32) + bf16 conversion ----------------

__global__ void ctl_prep_kernel(const float* __restrict__ x, const float* __restrict__ y,
                                unsigned short* __restrict__ xbf, unsigned short* __restrict__ ybf,
                                float* __restrict__ pos) {
    const int wave = threadIdx.x >> 5;
    const int lane = threadIdx.x & 31;
    const int row  = blockIdx.x * 8 + wave;

    const float4* xr = (const float4*)(x + (size_t)row * DDIM);
    const float4* yr = (const float4*)(y + (size_t)row * DDIM);
    ushort4* xbr = (ushort4*)(xbf + (size_t)row * DDIM);
    ushort4* ybr = (ushort4*)(ybf + (size_t)row * DDIM);

    float acc = 0.0f;
#pragma unroll
    for (int j = 0; j < 8; ++j) {
        const int q = lane + 32 * j;
        float4 xv = xr[q];
        float4 yv = yr[q];
        acc += xv.x * yv.x + xv.y * yv.y + xv.z * yv.z + xv.w * yv.w;
        xbr[q] = make_ushort4(f2bf(xv.x), f2bf(xv.y), f2bf(xv.z), f2bf(xv.w));
        ybr[q] = make_ushort4(f2bf(yv.x), f2bf(yv.y), f2bf(yv.z), f2bf(yv.w));
    }
#pragma unroll
    for (int m = 16; m >= 1; m >>= 1) acc += __shfl_xor(acc, m, 32);
    if (lane == 0) pos[row] = acc;
}

// ---------------- kernel 3: TDM-staged WMMA bf16 GEMM + masked argmax ----------------
// grid = (64, 64): one 128x128 output tile per workgroup; 8 waves arranged 4x2:
// each wave owns 32 rows x 64 cols = 2 A-frags x 4 B-frags -> 8 WMMAs per k-step.
// k-loop manually unrolled x2 over the double buffer: constant LDS addresses,
// unconditional mid-loop TDM issue, 12 ds_load_b128 then an 8-WMMA burst
// (sched_barrier keeps the loads from sinking into the WMMA sequence).

__launch_bounds__(256)
__global__ void ctl_sim_argmax_kernel(const unsigned short* __restrict__ xbf,
                                      const unsigned short* __restrict__ ybf,
                                      const float* __restrict__ pos,
                                      unsigned long long* __restrict__ packed) {
    __shared__ uint4 At[2][TILE_U4];
    __shared__ uint4 Bt[2][TILE_U4];

    const int lane  = threadIdx.x & 31;
    const int wave  = threadIdx.x >> 5;
    const int lr    = lane & 15;
    const int half  = lane >> 4;
    const int rowbase = blockIdx.x * 128;
    const int colbase = blockIdx.y * 128;
    const int rw = wave >> 1;                    // row group 0..3  -> 32 rows
    const int cw = wave & 1;                     // col group 0..1  -> 64 cols
    const int mrow = rw * 32;                    // wave-local row base
    const int mcol = cw * 64;                    // wave-local col base
    const int wave_u = __builtin_amdgcn_readfirstlane(threadIdx.x >> 5);  // uniform

    float pv[2][8];
#pragma unroll
    for (int r = 0; r < 2; ++r)
#pragma unroll
        for (int v = 0; v < 8; ++v)
            pv[r][v] = pos[rowbase + mrow + r * 16 + v + 8 * half];

    v8f c[2][4];
#pragma unroll
    for (int r = 0; r < 2; ++r)
#pragma unroll
        for (int t = 0; t < 4; ++t) c[r][t] = (v8f)0.0f;

    // stage tiles for K range [k, k+32) into buffer `buf` (A by wave0, B by wave1)
    auto stage = [&](int buf, int k) {
        if (wave_u == 0)
            tdm_tile_load(lds_off(&At[buf][0]), xbf + (size_t)rowbase * DDIM + k);
        if (wave_u == 1)
            tdm_tile_load(lds_off(&Bt[buf][0]), ybf + (size_t)colbase * DDIM + k);
    };

    // one K=32 step out of buffer `p`: 12 fragment loads, then 8-WMMA burst
    auto step = [&](int p) {
        // A-matrix 16x32 fragment (ISA 7.12.2): lanes 0-15 take K[0..7],K[16..23];
        // lanes 16-31 take K[8..15],K[24..31].  Row stride in LDS = 5 uint4 (80B).
        union { uint4 u[2]; v16bf v; } a[2], b[4];
#pragma unroll
        for (int r = 0; r < 2; ++r) {
            const int ar = (mrow + r * 16 + lr) * 5 + half;
            a[r].u[0] = At[p][ar];
            a[r].u[1] = At[p][ar + 2];
        }
#pragma unroll
        for (int t = 0; t < 4; ++t) {
            // B-matrix 32x16 fragment: lane = column; lanes 0-15 K[0..7],K[16..23],
            // lanes 16-31 K[8..15],K[24..31] (same order as A).
            const int br = (mcol + t * 16 + lr) * 5 + half;
            b[t].u[0] = Bt[p][br];
            b[t].u[1] = Bt[p][br + 2];
        }
#if __has_builtin(__builtin_amdgcn_sched_barrier)
        __builtin_amdgcn_sched_barrier(0);   // keep all DS loads ahead of the WMMA burst
#endif
#pragma unroll
        for (int t = 0; t < 4; ++t)
#pragma unroll
            for (int r = 0; r < 2; ++r)
                c[r][t] = __builtin_amdgcn_wmma_f32_16x16x32_bf16(
                    false, a[r].v, false, b[t].v, (short)0, c[r][t], false, false);
    };

    stage(0, 0);   // prologue

    for (int k0 = 0; k0 < DDIM; k0 += 64) {
        // phase 0: compute buffer 0, prefetch K[k0+32] into buffer 1
        if (wave_u <= 1) __builtin_amdgcn_s_wait_tensorcnt(0);
        __syncthreads();           // buf0 ready; all waves done reading buf1
        stage(1, k0 + 32);         // k0+32 <= 992 < DDIM always
        step(0);

        // phase 1: compute buffer 1, prefetch K[k0+64] into buffer 0
        if (wave_u <= 1) __builtin_amdgcn_s_wait_tensorcnt(0);
        __syncthreads();           // buf1 ready; all waves done reading buf0
        if (k0 + 64 < DDIM) stage(0, k0 + 64);
        step(1);
    }

    // C layout: (M,N) -> lane = N + 16*(M>=8), vgpr = M&7.
    float    bestv[2][8];
    unsigned besti[2][8];
#pragma unroll
    for (int r = 0; r < 2; ++r)
#pragma unroll
        for (int v = 0; v < 8; ++v) { bestv[r][v] = -1.0f; besti[r][v] = 0u; }

#pragma unroll
    for (int t = 0; t < 4; ++t) {
        const int col = colbase + mcol + t * 16 + lr;
#pragma unroll
        for (int r = 0; r < 2; ++r) {
#pragma unroll
            for (int v = 0; v < 8; ++v) {
                const int row = rowbase + mrow + r * 16 + v + 8 * half;
                float val = c[r][t][v];
                if (col == row) val = -1.0f;                       // diagonal mask
                else if (pv[r][v] - val > SEMI_HARD) val = -1.0f;  // semi-hard filter
                if (val > bestv[r][v]) { bestv[r][v] = val; besti[r][v] = (unsigned)col; }
            }
        }
    }

#pragma unroll
    for (int r = 0; r < 2; ++r) {
#pragma unroll
        for (int v = 0; v < 8; ++v) {
            float    bv = bestv[r][v];
            unsigned bi = besti[r][v];
#pragma unroll
            for (int m = 1; m < 16; m <<= 1) {
                float    ov = __shfl_xor(bv, m, 32);
                unsigned oi = __shfl_xor(bi, m, 32);
                if (ov > bv || (ov == bv && oi < bi)) { bv = ov; bi = oi; }
            }
            if (lr == 0)
                atomicMax(&packed[rowbase + mrow + r * 16 + v + 8 * half], pack_max(bv, bi));
        }
    }
}

// ---------------- kernel 4: negative_sim (fp32) + per-row hinge ----------------

__global__ void ctl_finalize_kernel(const float* __restrict__ x, const float* __restrict__ y,
                                    const float* __restrict__ pos,
                                    const unsigned long long* __restrict__ packed,
                                    float* __restrict__ loss_row) {
    const int wave = threadIdx.x >> 5;
    const int lane = threadIdx.x & 31;
    const int row  = blockIdx.x * 8 + wave;

    const unsigned idx = 0xFFFFFFFFu - (unsigned)(packed[row] & 0xFFFFFFFFull);

    const float4* xr = (const float4*)(x + (size_t)row * DDIM);
    const float4* yr = (const float4*)(y + (size_t)idx * DDIM);
    float acc = 0.0f;
#pragma unroll
    for (int j = 0; j < 8; ++j) {
        const int q = lane + 32 * j;
        float4 a = xr[q];
        float4 b = yr[q];
        acc += a.x * b.x + a.y * b.y + a.z * b.z + a.w * b.w;
    }
#pragma unroll
    for (int m = 16; m >= 1; m >>= 1) acc += __shfl_xor(acc, m, 32);
    if (lane == 0) {
        float l = MARGIN - pos[row] + acc;
        loss_row[row] = l > 0.0f ? l : 0.0f;
    }
}

// ---------------- kernel 5: deterministic mean reduction ----------------

__global__ void ctl_reduce_kernel(const float* __restrict__ loss_row, float* __restrict__ out) {
    __shared__ float sm[256];
    float acc = 0.0f;
    for (int i = threadIdx.x; i < BDIM; i += 256) acc += loss_row[i];
    sm[threadIdx.x] = acc;
    __syncthreads();
    for (int s = 128; s > 0; s >>= 1) {
        if (threadIdx.x < s) sm[threadIdx.x] += sm[threadIdx.x + s];
        __syncthreads();
    }
    if (threadIdx.x == 0) out[0] = sm[0] * (1.0f / (float)BDIM);
}

// ---------------- launch ----------------

extern "C" void kernel_launch(void* const* d_in, const int* in_sizes, int n_in,
                              void* d_out, int out_size, void* d_ws, size_t ws_size,
                              hipStream_t stream) {
    const float* x = (const float*)d_in[0];
    const float* y = (const float*)d_in[1];
    float* out = (float*)d_out;

    // workspace: xbf(16MB) | ybf(16MB) | pos(32KB) | packed(64KB) | loss_row(32KB)
    unsigned short* xbf = (unsigned short*)d_ws;
    unsigned short* ybf = xbf + (size_t)BDIM * DDIM;
    float* pos = (float*)(ybf + (size_t)BDIM * DDIM);
    unsigned long long* packed = (unsigned long long*)(pos + BDIM);
    float* loss_row = (float*)(packed + BDIM);

    hipLaunchKernelGGL(ctl_init_kernel, dim3(BDIM / 256), dim3(256), 0, stream, packed);
    hipLaunchKernelGGL(ctl_prep_kernel, dim3(BDIM / 8), dim3(256), 0, stream,
                       x, y, xbf, ybf, pos);
    hipLaunchKernelGGL(ctl_sim_argmax_kernel, dim3(BDIM / 128, BDIM / 128), dim3(256), 0, stream,
                       xbf, ybf, pos, packed);
    hipLaunchKernelGGL(ctl_finalize_kernel, dim3(BDIM / 8), dim3(256), 0, stream,
                       x, y, pos, packed, loss_row);
    hipLaunchKernelGGL(ctl_reduce_kernel, dim3(1), dim3(256), 0, stream, loss_row, out);
}
